// SelfAttentionLayer_52905407152735
// MI455X (gfx1250) — matrix operand order repaired and checked by reading, not compiled, not run on hardware
//
#include <hip/hip_runtime.h>
#include <hip/hip_bf16.h>
#include <math.h>

#define N_NODES 50000
#define N_EDGES 400000
#define NODE_DIM 128
#define EDGE_DIM 64
#define HEADS 8

typedef __attribute__((ext_vector_type(16))) _Float16 v16h;
typedef __attribute__((ext_vector_type(8)))  float    v8f;

__device__ __forceinline__ float silu_f(float x) {
    return x * (1.0f / (1.0f + __expf(-x)));
}

__device__ __forceinline__ void atomicMaxF(float* addr, float val) {
    unsigned int* ua = (unsigned int*)addr;
    unsigned int old = *ua;
    while (true) {
        float f = __uint_as_float(old);
        if (f >= val) break;
        unsigned int assumed = old;
        old = atomicCAS(ua, assumed, __float_as_uint(val));
        if (old == assumed) break;
    }
}

// Async copy of 32 bytes/thread: global -> LDS, two B128 ops sharing one
// address setup (INST_OFFSET applies to both LDS and global sides).
__device__ __forceinline__ void async_copy_32B(unsigned lds_byte_off,
                                               const void* gaddr) {
    asm volatile("global_load_async_to_lds_b128 %0, %1, off\n\t"
                 "global_load_async_to_lds_b128 %0, %1, off offset:16"
                 :: "v"(lds_byte_off), "v"(gaddr) : "memory");
}
__device__ __forceinline__ void async_copy_16B(unsigned lds_byte_off,
                                               const void* gaddr) {
    asm volatile("global_load_async_to_lds_b128 %0, %1, off"
                 :: "v"(lds_byte_off), "v"(gaddr) : "memory");
}
__device__ __forceinline__ void wait_async() {
    asm volatile("s_wait_asynccnt 0x0" ::: "memory");
}

// Load B fragment for 16x16x32 f16 WMMA from a transposed (column-major) f16
// weight: wT[n * Kdim + k].  Lane l -> column n = l%16; half = l/16;
// VGPR v holds K = 16*(v/4) + 8*half + 2*(v%4), +1.
__device__ __forceinline__ v16h load_bfrag(const _Float16* __restrict__ wT,
                                           int Kdim, int n, int half, int kb) {
    v16h b;
    const _Float16* base = wT + n * Kdim + kb * 32 + (half << 3);
#pragma unroll
    for (int v = 0; v < 8; ++v) {
        b[2 * v]     = base[2 * v + ((v >> 2) << 3)];
        b[2 * v + 1] = base[2 * v + 1 + ((v >> 2) << 3)];
    }
    return b;
}

// Build an A fragment (16x32 f16, one K-block) from an f32 tile in LDS,
// tile layout row-major [16][ldm].
__device__ __forceinline__ v16h lds_afrag_f32(const float* __restrict__ tile,
                                              int ldm, int m, int half, int kb) {
    v16h a;
#pragma unroll
    for (int v = 0; v < 8; ++v) {
        int k = kb * 32 + ((v >> 2) << 4) + (half << 3) + ((v & 3) << 1);
        a[2 * v]     = (_Float16)tile[m * ldm + k];
        a[2 * v + 1] = (_Float16)tile[m * ldm + k + 1];
    }
    return a;
}

// ---------------------------------------------------------------------------
// Kernel W: transpose f32 weight [K x Nc] (row-major) -> f16 wT[n*K + k]
// ---------------------------------------------------------------------------
__global__ void transpose_f16_kernel(const float* __restrict__ src,
                                     _Float16* __restrict__ dst,
                                     int K, int Nc) {
    int i = blockIdx.x * blockDim.x + threadIdx.x;
    if (i >= K * Nc) return;
    int k = i / Nc;
    int n = i % Nc;
    dst[n * K + k] = (_Float16)src[k * Nc + n];
}

// ---------------------------------------------------------------------------
// Kernel I: init segment-softmax stats
// ---------------------------------------------------------------------------
__global__ void init_stats_kernel(float* __restrict__ Amax,
                                  float* __restrict__ Asum,
                                  float* __restrict__ Deg) {
    int t = blockIdx.x * blockDim.x + threadIdx.x;
    const int NH = N_NODES * HEADS;
    if (t < NH)               Amax[t] = -INFINITY;
    else if (t < 2 * NH)      Asum[t - NH] = 0.0f;
    else if (t < 2 * NH + N_NODES) Deg[t - 2 * NH] = 0.0f;
}

// ---------------------------------------------------------------------------
// Kernel N: node projections Q, K, V = silu(h@mw1+mb1)@mw2+mb2
// one block = 16 nodes, 8 waves; wave w owns output columns [16w,16w+16)
// A-tile staged once per block via async global->LDS copy.
// ---------------------------------------------------------------------------
__global__ __launch_bounds__(256) void node_proj_kernel(
    const float* __restrict__ h,
    const _Float16* __restrict__ wqT, const _Float16* __restrict__ wkT,
    const _Float16* __restrict__ mw1T, const _Float16* __restrict__ mw2T,
    const float* __restrict__ bq, const float* __restrict__ bk,
    const float* __restrict__ mb1, const float* __restrict__ mb2,
    float* __restrict__ Q, float* __restrict__ Kf, float* __restrict__ V) {
    __shared__ float    htile[16 * NODE_DIM];   // 8 KB
    __shared__ _Float16 zlds[16 * NODE_DIM];    // 4 KB

    const int m0   = blockIdx.x * 16;
    const int t    = threadIdx.x;
    const int lane = t & 31;
    const int w    = t >> 5;
    const int n0   = w * 16;
    const int nl   = lane & 15;
    const int half = lane >> 4;

    // async stage: 16*128 f32 = 8192 B, 32 B per thread
    async_copy_32B((unsigned)(t * 32), (const void*)(h + (size_t)m0 * NODE_DIM + t * 8));
    __builtin_prefetch(wqT + n0 * NODE_DIM, 0, 0);   // warm B fragments in L2
    wait_async();
    __syncthreads();

    // A fragments: 16x128 tile, 4 K-blocks of 32
    v16h afrag[4];
#pragma unroll
    for (int kb = 0; kb < 4; ++kb) afrag[kb] = lds_afrag_f32(htile, NODE_DIM, nl, half, kb);

    // Q projection
    {
        float bias = bq[n0 + nl];
        v8f c;
#pragma unroll
        for (int d = 0; d < 8; ++d) c[d] = bias;
#pragma unroll
        for (int kb = 0; kb < 4; ++kb)
            c = __builtin_amdgcn_wmma_f32_16x16x32_f16(
                    false, afrag[kb], false, load_bfrag(wqT, 128, n0 + nl, half, kb),
                    (short)0, c, false, false);
#pragma unroll
        for (int d = 0; d < 8; ++d)
            Q[(m0 + d + 8 * half) * NODE_DIM + n0 + nl] = c[d];
    }
    // K projection
    {
        float bias = bk[n0 + nl];
        v8f c;
#pragma unroll
        for (int d = 0; d < 8; ++d) c[d] = bias;
#pragma unroll
        for (int kb = 0; kb < 4; ++kb)
            c = __builtin_amdgcn_wmma_f32_16x16x32_f16(
                    false, afrag[kb], false, load_bfrag(wkT, 128, n0 + nl, half, kb),
                    (short)0, c, false, false);
#pragma unroll
        for (int d = 0; d < 8; ++d)
            Kf[(m0 + d + 8 * half) * NODE_DIM + n0 + nl] = c[d];
    }
    // hidden z = silu(h@mw1 + mb1) -> LDS (f16)
    {
        float bias = mb1[n0 + nl];
        v8f c;
#pragma unroll
        for (int d = 0; d < 8; ++d) c[d] = bias;
#pragma unroll
        for (int kb = 0; kb < 4; ++kb)
            c = __builtin_amdgcn_wmma_f32_16x16x32_f16(
                    false, afrag[kb], false, load_bfrag(mw1T, 128, n0 + nl, half, kb),
                    (short)0, c, false, false);
#pragma unroll
        for (int d = 0; d < 8; ++d)
            zlds[(d + 8 * half) * NODE_DIM + n0 + nl] = (_Float16)silu_f(c[d]);
    }
    __syncthreads();
    // V = z @ mw2 + mb2   (A fragments from LDS, already f16)
    {
        v16h zfrag[4];
#pragma unroll
        for (int kb = 0; kb < 4; ++kb) {
#pragma unroll
            for (int v = 0; v < 8; ++v) {
                int k = kb * 32 + ((v >> 2) << 4) + (half << 3) + ((v & 3) << 1);
                zfrag[kb][2 * v]     = zlds[nl * NODE_DIM + k];
                zfrag[kb][2 * v + 1] = zlds[nl * NODE_DIM + k + 1];
            }
        }
        float bias = mb2[n0 + nl];
        v8f c;
#pragma unroll
        for (int d = 0; d < 8; ++d) c[d] = bias;
#pragma unroll
        for (int kb = 0; kb < 4; ++kb)
            c = __builtin_amdgcn_wmma_f32_16x16x32_f16(
                    false, zfrag[kb], false, load_bfrag(mw2T, 128, n0 + nl, half, kb),
                    (short)0, c, false, false);
#pragma unroll
        for (int d = 0; d < 8; ++d)
            V[(m0 + d + 8 * half) * NODE_DIM + n0 + nl] = c[d];
    }
}

// ---------------------------------------------------------------------------
// Kernel E1: per 16-edge tile -> re = silu(t_ij@wre+bre) via WMMA,
// a_ij[e][head] = sum_d q_i*k_j*re ; store logits + atomic segment max
// wave w == head w (16 feature cols per head)
// ---------------------------------------------------------------------------
__global__ __launch_bounds__(256) void edge_logits_kernel(
    const float* __restrict__ t_ij, const int* __restrict__ ei,
    const _Float16* __restrict__ wreT, const float* __restrict__ bre,
    const float* __restrict__ Q, const float* __restrict__ Kf,
    float* __restrict__ Aij, float* __restrict__ Amax) {
    __shared__ float ttile[16 * EDGE_DIM];   // 4 KB
    const int e0   = blockIdx.x * 16;
    const int t    = threadIdx.x;
    const int lane = t & 31;
    const int w    = t >> 5;   // head
    const int n0   = w * 16;
    const int nl   = lane & 15;
    const int half = lane >> 4;

    // async stage: 16*64 f32 = 4096 B, 16 B per thread
    async_copy_16B((unsigned)(t * 16), (const void*)(t_ij + (size_t)e0 * EDGE_DIM + t * 4));
    __builtin_prefetch(wreT + n0 * EDGE_DIM, 0, 0);
    wait_async();
    __syncthreads();

    // A fragments: 16x64 tile of t_ij, 2 K-blocks
    v16h afrag[2];
#pragma unroll
    for (int kb = 0; kb < 2; ++kb) afrag[kb] = lds_afrag_f32(ttile, EDGE_DIM, nl, half, kb);

    float bias = bre[n0 + nl];
    v8f c;
#pragma unroll
    for (int d = 0; d < 8; ++d) c[d] = bias;
#pragma unroll
    for (int kb = 0; kb < 2; ++kb)
        c = __builtin_amdgcn_wmma_f32_16x16x32_f16(
                false, afrag[kb], false, load_bfrag(wreT, 64, n0 + nl, half, kb),
                (short)0, c, false, false);

    // per-lane partial products, then reduce over the 16 feature lanes
    float p[8];
#pragma unroll
    for (int d = 0; d < 8; ++d) {
        int e  = e0 + d + 8 * half;
        int nj = ei[e];
        int ni = ei[N_EDGES + e];
        float re = silu_f(c[d]);
        float qv = Q[ni * NODE_DIM + n0 + nl];
        float kv = Kf[nj * NODE_DIM + n0 + nl];
        p[d] = qv * kv * re;
    }
#pragma unroll
    for (int mask = 1; mask < 16; mask <<= 1) {
#pragma unroll
        for (int d = 0; d < 8; ++d) p[d] += __shfl_xor(p[d], mask, 32);
    }
    if (nl == 0) {
#pragma unroll
        for (int d = 0; d < 8; ++d) {
            int e  = e0 + d + 8 * half;
            int ni = ei[N_EDGES + e];
            Aij[e * HEADS + w] = p[d];
            atomicMaxF(&Amax[ni * HEADS + w], p[d]);
        }
    }
}

// ---------------------------------------------------------------------------
// Kernel E2: exp(a - max), segment sum, degree count
// ---------------------------------------------------------------------------
__global__ void edge_softmax_kernel(const int* __restrict__ ei,
                                    float* __restrict__ Aij,
                                    const float* __restrict__ Amax,
                                    float* __restrict__ Asum,
                                    float* __restrict__ Deg) {
    int t = blockIdx.x * blockDim.x + threadIdx.x;
    if (t >= N_EDGES * HEADS) return;
    int e  = t >> 3;
    int hh = t & 7;
    int ni = ei[N_EDGES + e];
    float m = Amax[ni * HEADS + hh];
    if (!(m > -1e30f && m < 1e30f)) m = 0.0f;
    float ex = __expf(Aij[t] - m);
    Aij[t] = ex;
    atomicAdd(&Asum[ni * HEADS + hh], ex);
    if (hh == 0) atomicAdd(&Deg[ni], 1.0f);
}

// ---------------------------------------------------------------------------
// Kernel E3: sea = a_i * v_j (gather, normalize) -> LDS f16 -> WMMA @ wc + bc
// ---------------------------------------------------------------------------
__global__ __launch_bounds__(256) void edge_out_kernel(
    const int* __restrict__ ei, const float* __restrict__ Aij,
    const float* __restrict__ Asum, const float* __restrict__ Deg,
    const float* __restrict__ V, const _Float16* __restrict__ wcT,
    const float* __restrict__ bc, float* __restrict__ out) {
    __shared__ _Float16 slds[16 * NODE_DIM];
    const int e0 = blockIdx.x * 16;
    const int t  = threadIdx.x;

    // stage sea tile: thread t -> edge row t/16, 8 consecutive cols
    // (each 8-col span lies inside a single head -> one weight per thread)
    {
        int r  = t >> 4;
        int cs = (t & 15) * 8;
        int hh = cs >> 4;
        int e  = e0 + r;
        int ni = ei[N_EDGES + e];
        int nj = ei[e];
        float dnorm = sqrtf(Deg[ni]) * 0.25f;   // sqrt(deg)/sqrt(16)
        float a = Aij[e * HEADS + hh] / Asum[ni * HEADS + hh] * dnorm;
        const float4* vr = (const float4*)(V + (size_t)nj * NODE_DIM + cs);
        float4 x0 = vr[0];
        float4 x1 = vr[1];
        _Float16* s = slds + r * NODE_DIM + cs;
        s[0] = (_Float16)(a * x0.x);  s[1] = (_Float16)(a * x0.y);
        s[2] = (_Float16)(a * x0.z);  s[3] = (_Float16)(a * x0.w);
        s[4] = (_Float16)(a * x1.x);  s[5] = (_Float16)(a * x1.y);
        s[6] = (_Float16)(a * x1.z);  s[7] = (_Float16)(a * x1.w);
    }
    __syncthreads();

    const int lane = t & 31;
    const int w    = t >> 5;
    const int n0   = w * 16;
    const int nl   = lane & 15;
    const int half = lane >> 4;

    v16h afrag[4];
#pragma unroll
    for (int kb = 0; kb < 4; ++kb) {
#pragma unroll
        for (int v = 0; v < 8; ++v) {
            int k = kb * 32 + ((v >> 2) << 4) + (half << 3) + ((v & 3) << 1);
            afrag[kb][2 * v]     = slds[nl * NODE_DIM + k];
            afrag[kb][2 * v + 1] = slds[nl * NODE_DIM + k + 1];
        }
    }
    float bias = bc[n0 + nl];
    v8f c;
#pragma unroll
    for (int d = 0; d < 8; ++d) c[d] = bias;
#pragma unroll
    for (int kb = 0; kb < 4; ++kb)
        c = __builtin_amdgcn_wmma_f32_16x16x32_f16(
                false, afrag[kb], false, load_bfrag(wcT, 128, n0 + nl, half, kb),
                (short)0, c, false, false);
#pragma unroll
    for (int d = 0; d < 8; ++d)
        out[(e0 + d + 8 * half) * NODE_DIM + n0 + nl] = c[d];
}

// ---------------------------------------------------------------------------
extern "C" void kernel_launch(void* const* d_in, const int* in_sizes, int n_in,
                              void* d_out, int out_size, void* d_ws, size_t ws_size,
                              hipStream_t stream) {
    const float* h    = (const float*)d_in[0];
    const float* t_ij = (const float*)d_in[1];
    const int*   ei   = (const int*)d_in[2];
    const float* wq   = (const float*)d_in[3];
    const float* bq   = (const float*)d_in[4];
    const float* wk   = (const float*)d_in[5];
    const float* bk   = (const float*)d_in[6];
    const float* wre  = (const float*)d_in[7];
    const float* bre  = (const float*)d_in[8];
    const float* mw1  = (const float*)d_in[9];
    const float* mb1  = (const float*)d_in[10];
    const float* mw2  = (const float*)d_in[11];
    const float* mb2  = (const float*)d_in[12];
    const float* wc   = (const float*)d_in[13];
    const float* bc   = (const float*)d_in[14];
    float* out = (float*)d_out;

    // workspace carve-up (floats)
    float* w = (float*)d_ws;
    size_t o = 0;
    float* Q    = w + o; o += (size_t)N_NODES * NODE_DIM;
    float* Kf   = w + o; o += (size_t)N_NODES * NODE_DIM;
    float* Vv   = w + o; o += (size_t)N_NODES * NODE_DIM;
    float* Aij  = w + o; o += (size_t)N_EDGES * HEADS;
    float* Amax = w + o; o += (size_t)N_NODES * HEADS;
    float* Asum = w + o; o += (size_t)N_NODES * HEADS;
    float* Deg  = w + o; o += (size_t)N_NODES;
    _Float16* f16base = (_Float16*)(w + o);
    _Float16* wqT  = f16base;           // 128*128
    _Float16* wkT  = wqT  + 128 * 128;
    _Float16* mw1T = wkT  + 128 * 128;
    _Float16* mw2T = mw1T + 128 * 128;
    _Float16* wcT  = mw2T + 128 * 128;
    _Float16* wreT = wcT  + 128 * 128;  // 128*64

    // W: weight transposes -> f16 column-major
    transpose_f16_kernel<<<(128 * 128 + 255) / 256, 256, 0, stream>>>(wq,  wqT,  128, 128);
    transpose_f16_kernel<<<(128 * 128 + 255) / 256, 256, 0, stream>>>(wk,  wkT,  128, 128);
    transpose_f16_kernel<<<(128 * 128 + 255) / 256, 256, 0, stream>>>(mw1, mw1T, 128, 128);
    transpose_f16_kernel<<<(128 * 128 + 255) / 256, 256, 0, stream>>>(mw2, mw2T, 128, 128);
    transpose_f16_kernel<<<(128 * 128 + 255) / 256, 256, 0, stream>>>(wc,  wcT,  128, 128);
    transpose_f16_kernel<<<(64 * 128 + 255) / 256, 256, 0, stream>>>(wre, wreT,  64, 128);

    // I: init stats
    {
        int tot = 2 * N_NODES * HEADS + N_NODES;
        init_stats_kernel<<<(tot + 255) / 256, 256, 0, stream>>>(Amax, Asum, Deg);
    }
    // N: node projections (50000/16 = 3125 tiles)
    node_proj_kernel<<<N_NODES / 16, 256, 0, stream>>>(
        h, wqT, wkT, mw1T, mw2T, bq, bk, mb1, mb2, Q, Kf, Vv);
    // E1: edge logits + segment max (400000/16 = 25000 tiles)
    edge_logits_kernel<<<N_EDGES / 16, 256, 0, stream>>>(
        t_ij, ei, wreT, bre, Q, Kf, Aij, Amax);
    // E2: exp + segment sum + degree
    edge_softmax_kernel<<<(N_EDGES * HEADS) / 256, 256, 0, stream>>>(
        ei, Aij, Amax, Asum, Deg);
    // E3: weighted value gather + output GEMM
    edge_out_kernel<<<N_EDGES / 16, 256, 0, stream>>>(
        ei, Aij, Asum, Deg, Vv, wcT, bc, out);
}
